// con_estimator_4D_47253230190960
// MI455X (gfx1250) — matrix operand order repaired and checked by reading, not compiled, not run on hardware
//
#include <hip/hip_runtime.h>
#include <stdint.h>

// CDNA5 / gfx1250: wave32, WMMA 16x16x32 bf16 (f32 accumulate), TDM staging.
typedef __attribute__((ext_vector_type(16))) __bf16 v16bf;
typedef __attribute__((ext_vector_type(8)))  float  v8f;

typedef __attribute__((ext_vector_type(4))) unsigned int tdm_v4u;
typedef __attribute__((ext_vector_type(8))) int          tdm_v8i;
typedef __attribute__((ext_vector_type(4))) int          tdm_v4i;

union Frag {
    uint4          q[2];
    unsigned short h[16];
    v16bf          v;
};

__device__ __forceinline__ unsigned short f32_to_bf16_bits(float f) {
    uint32_t u = __float_as_uint(f);
    uint32_t r = u + 0x7FFFu + ((u >> 16) & 1u);   // round-to-nearest-even
    return (unsigned short)(r >> 16);
}

__global__ __launch_bounds__(256) void cvt_f32_bf16(const float* __restrict__ in,
                                                    unsigned short* __restrict__ out,
                                                    int n) {
    int i = blockIdx.x * 256 + threadIdx.x;
    int stride = gridDim.x * 256;
    for (; i < n; i += stride) out[i] = f32_to_bf16_bits(in[i]);
}

// Tensor Data Mover: 2D tile load, 16 lines of 256 bf16 elements, with LDS
// padding (skip 8 DWORDs after each 8 stored) so every 16-element T-row lands
// in its own 64B slot -> directly produces the zero-padded row-major A tiles.
__device__ __forceinline__ void tdm_load_slab(uint32_t lds_addr,
                                              const unsigned short* gptr,
                                              uint64_t line_stride_elems) {
    const uint64_t ga = (uint64_t)(uintptr_t)gptr;
    // group0: count=1, lds_addr, global_addr, type=2 ("image")
    tdm_v4u g0 = { 1u,
                   lds_addr,
                   (uint32_t)ga,
                   (uint32_t)((ga >> 32) & 0x01FFFFFFu) | (2u << 30) };
    // group1: data_size=2B(code1), pad_enable, pad_interval=8DW(code2),
    //         pad_amount=8DW(code7); tensor_dim0=256, tile_dim0=256,
    //         tensor_dim1=16, tile_dim1=16, tile_dim2=0, dim0_stride=batch.
    const uint32_t d0   = (1u << 16) | (1u << 20) | (2u << 22) | (7u << 25);
    const uint32_t td0  = 256, td1 = 16, tile0 = 256, tile1 = 16;
    tdm_v8i g1 = { (int)d0,
                   (int)((td0 & 0xFFFFu) << 16),
                   (int)(((td0 >> 16) & 0xFFFFu) | ((td1 & 0xFFFFu) << 16)),
                   (int)(((td1 >> 16) & 0xFFFFu) | ((tile0 & 0xFFFFu) << 16)),
                   (int)(tile1 & 0xFFFFu),
                   (int)(uint32_t)(line_stride_elems & 0xFFFFFFFFu),
                   (int)(uint32_t)((line_stride_elems >> 32) & 0xFFFFu),
                   0 };
    tdm_v4i gz = { 0, 0, 0, 0 };
#if defined(__clang__) && (__clang_major__ >= 23)
    tdm_v8i gz8 = { 0, 0, 0, 0, 0, 0, 0, 0 };
    __builtin_amdgcn_tensor_load_to_lds(g0, g1, gz, gz, gz8, 0);
#else
    __builtin_amdgcn_tensor_load_to_lds(g0, g1, gz, gz, 0);
#endif
}

// Banded-kernel GEMM 4D conv layer.
// out[b,t] over (d_o,h_o,w_o): M=16 batches, N=16 t-positions, K=32 band.
// Valid (kd,kh) taps form a RECTANGLE [kdLo,kdHi)x[khLo,khHi) -> plain counters.
template<int Kd,int Kh,int Kw,int Kt,int PD,int PH,int PW,int PT,
         int ID,int IH,int OD,int OH,int ACT>
__global__ __launch_bounds__(256)
void conv4d_wmma(const unsigned short* __restrict__ X,
                 const unsigned short* __restrict__ Wg,
                 const float* __restrict__ bias,
                 unsigned short* __restrict__ Ybf,
                 float* __restrict__ Yf)
{
    constexpr int O = (PT + 1) & ~1;     // band origin, even => DWORD-aligned LDS
    // raw[buf][b'][w_i] : 64B row-major padded k-row (k in [O, O+16) holds data)
    __shared__ __align__(16) unsigned short rawA[2][8192];   // 2 x 16 KiB
    __shared__ __align__(16) unsigned short bFrag[Kw * 16 * 32];

    const int tid  = threadIdx.x;
    const int lane = tid & 31;
    const int wave = tid >> 5;
    const int hf   = lane >> 4;
    const int m    = lane & 15;          // A-row (batch) / B-col (t)

    const int bx  = blockIdx.x;
    const int dh  = bx % (OD * OH);
    const int bt  = bx / (OD * OH);
    const int d_o = dh / OH;
    const int h_o = dh % OH;

    // ---- pre-zero raw slabs once: TDM pad regions stay zero forever --------
    {
        uint4 z; z.x = z.y = z.z = z.w = 0;
        uint4* p = (uint4*)&rawA[0][0];
        for (int i = tid; i < (int)(sizeof(rawA) / 16); i += 256) p[i] = z;
    }
    __syncthreads();

    // valid tap rectangle: d_i = d_o + kd - PD in [0,ID) <=> kd in [kdLo,kdHi)
    const int kdLo = (PD - d_o) > 0 ? (PD - d_o) : 0;
    const int kdHi = (PD - d_o + ID) < Kd ? (PD - d_o + ID) : Kd;
    const int khLo = (PH - h_o) > 0 ? (PH - h_o) : 0;
    const int khHi = (PH - h_o + IH) < Kh ? (PH - h_o + IH) : Kh;

    auto issueTdm = [&](int kd, int kh, int buf) {
        const int d_i = d_o + kd - PD, h_i = h_o + kh - PH;
        const unsigned short* g =
            X + ((size_t)((bt * 16) * ID + d_i) * IH + h_i) * 256;
        const uint32_t lds =
            (uint32_t)(uintptr_t)&rawA[buf][0] + (uint32_t)(O * 2);
        tdm_load_slab(lds, g, (uint64_t)ID * IH * 256);
    };
    auto loadA = [&](Frag& f, const unsigned short* rb, int w_i) {
        if (w_i >= 0 && w_i < 16) {
            const unsigned short* p = rb + (m * 16 + w_i) * 32 + hf * 8;
            f.q[0] = *(const uint4*)p;
            f.q[1] = *(const uint4*)(p + 16);
        } else {                          // w-halo: zero fragment (wave-uniform)
            uint4 z; z.x = z.y = z.z = z.w = 0;
            f.q[0] = z; f.q[1] = z;
        }
    };

    int kd = kdLo, kh = khLo;
    int buf = 0;
    if (wave == 0) issueTdm(kd, kh, buf);

    v8f acc0 = {};
    v8f acc1 = {};
    const int w_o0 = wave * 2;

    while (kd < kdHi) {
        // advance pipeline state (plain counters, no div/mod)
        int kdN = kd, khN = kh + 1;
        if (khN == khHi) { khN = khLo; ++kdN; }
        const bool hasNext = (kdN < kdHi);

        // ---- build banded B rows (SCATTER form: all weight loads are
        // unconditional -> batched, no per-element exec-mask/waits).
        // bFrag[kw][n][k] with row[k] = w[k-n-O+PT]  <=>  k = n+(O-PT)+kt.
        const unsigned short* wbase = Wg + ((size_t)kd * Kh + kh) * (Kw * Kt);
        for (int r = tid; r < Kw * 16; r += 256) {
            const int kw = r >> 4, n = r & 15;
            const unsigned short* wrow = wbase + kw * Kt;
            __builtin_prefetch(wrow, 0, 3);          // global_prefetch_b8
            unsigned short wreg[Kt];
#pragma unroll
            for (int kt = 0; kt < Kt; ++kt) wreg[kt] = wrow[kt];
            uint4 z; z.x = z.y = z.z = z.w = 0;
            uint4* dst = (uint4*)&bFrag[r * 32];
            dst[0] = z; dst[1] = z; dst[2] = z; dst[3] = z;
            unsigned short* row = &bFrag[r * 32 + n + (O - PT)];
#pragma unroll
            for (int kt = 0; kt < Kt; ++kt) row[kt] = wreg[kt];
        }

        __builtin_amdgcn_s_wait_tensorcnt(0);        // raw[buf] DMA complete
        __syncthreads();                             // publish raw[buf] + bFrag

        if (hasNext && wave == 0) issueTdm(kdN, khN, buf ^ 1);  // overlap DMA

        // ---- hot loop: rotated A fragments, pipelined LDS loads ------------
        const unsigned short* rb = &rawA[buf][0];
        Frag aC, aN;
        loadA(aC, rb, w_o0 - PW);
        loadA(aN, rb, w_o0 + 1 - PW);
#pragma unroll
        for (int kw = 0; kw < Kw; ++kw) {
            Frag bc;
            {   // B fragment: lane n=m, chunks at k = hf*8 + [0..7] and 16 + hf*8 + [0..7]
                const unsigned short* p = &bFrag[(kw * 16 + m) * 32 + hf * 8];
                bc.q[0] = *(const uint4*)p;
                bc.q[1] = *(const uint4*)(p + 16);
            }
            Frag aN2;
            if (kw + 1 < Kw) loadA(aN2, rb, w_o0 + kw + 2 - PW);  // prefetch
            acc0 = __builtin_amdgcn_wmma_f32_16x16x32_bf16(
                false, aC.v, false, bc.v, (short)0, acc0, false, false);
            acc1 = __builtin_amdgcn_wmma_f32_16x16x32_bf16(
                false, aN.v, false, bc.v, (short)0, acc1, false, false);
            aC = aN;
            if (kw + 1 < Kw) aN = aN2;
        }

        kd = kdN; kh = khN;
        buf ^= 1;
        __syncthreads();                 // compute done before bFrag rebuild
    }

    // ---- epilogue: C/D layout — lane n=L%16, VGPR r holds M = r + 8*(L/16) -
    const float bv = bias[0];
#pragma unroll
    for (int j = 0; j < 2; ++j) {
        const int w_o = w_o0 + j;
        const v8f  c  = j ? acc1 : acc0;
#pragma unroll
        for (int rr = 0; rr < 8; ++rr) {
            const int bp = rr + 8 * hf;
            const int b  = bt * 16 + bp;
            const int oi = (((b * OD + d_o) * OH + h_o) * 16 + w_o) * 16 + m;
            float v = c[rr] + bv;
            if (ACT == 0) {
                v = v > 0.0f ? v : 0.0f;
                Ybf[oi] = f32_to_bf16_bits(v);
            } else {
                Yf[oi] = 1.0f / (1.0f + __expf(-v));
            }
        }
    }
}

// -------------------------- host side ---------------------------------------
// Workspace layout (bytes); requires ~51 MB of d_ws.
static const size_t OFF_XB   = 0;                       // 16,777,216 bf16 = 32 MiB
static const size_t OFF_BUFA = 33554432;                //  4,194,304 bf16 =  8 MiB
static const size_t OFF_BUFB = OFF_BUFA + 8388608;
static const size_t OFF_W    = OFF_BUFB + 8388608;      // bf16 weights

extern "C" void kernel_launch(void* const* d_in, const int* in_sizes, int n_in,
                              void* d_out, int out_size, void* d_ws, size_t ws_size,
                              hipStream_t stream) {
    (void)in_sizes; (void)n_in; (void)out_size; (void)ws_size;

    const float* corr = (const float*)d_in[0];
    const float* w1 = (const float*)d_in[1];  const float* b1 = (const float*)d_in[2];
    const float* w2 = (const float*)d_in[3];  const float* b2 = (const float*)d_in[4];
    const float* w3 = (const float*)d_in[5];  const float* b3 = (const float*)d_in[6];
    const float* w4 = (const float*)d_in[7];  const float* b4 = (const float*)d_in[8];
    const float* w5 = (const float*)d_in[9];  const float* b5 = (const float*)d_in[10];

    char* ws = (char*)d_ws;
    unsigned short* Xb   = (unsigned short*)(ws + OFF_XB);
    unsigned short* bufA = (unsigned short*)(ws + OFF_BUFA);
    unsigned short* bufB = (unsigned short*)(ws + OFF_BUFB);
    unsigned short* wgt  = (unsigned short*)(ws + OFF_W);
    unsigned short* w1b = wgt;                 // 23409 -> pad 23424
    unsigned short* w2b = w1b + 23424;         //  2401 -> pad 2432
    unsigned short* w3b = w2b + 2432;          //   625 -> pad 640
    unsigned short* w4b = w3b + 640;           //    81 -> pad 96
    unsigned short* w5b = w4b + 96;            //    64

    // bf16 conversions (everything is L2-resident afterwards: << 192 MB L2)
    const int nX = 256 * 16 * 16 * 16 * 16;
    cvt_f32_bf16<<<8192, 256, 0, stream>>>(corr, Xb, nX);
    cvt_f32_bf16<<<92,   256, 0, stream>>>(w1, w1b, 23409);
    cvt_f32_bf16<<<10,   256, 0, stream>>>(w2, w2b, 2401);
    cvt_f32_bf16<<<3,    256, 0, stream>>>(w3, w3b, 625);
    cvt_f32_bf16<<<1,    256, 0, stream>>>(w4, w4b, 81);
    cvt_f32_bf16<<<1,    256, 0, stream>>>(w5, w5b, 64);

    // Layer 1: (17,17,9,9) pad 4, in (16,16,16,16) -> out (8,8,16,16)
    conv4d_wmma<17,17,9,9, 4,4,4,4, 16,16, 8,8, 0>
        <<<16 * 8 * 8, 256, 0, stream>>>(Xb, w1b, b1, bufA, nullptr);
    // Layer 2: (7,7,7,7) pad 3
    conv4d_wmma<7,7,7,7, 3,3,3,3, 8,8, 8,8, 0>
        <<<16 * 8 * 8, 256, 0, stream>>>(bufA, w2b, b2, bufB, nullptr);
    // Layer 3: (5,5,5,5) pad 2
    conv4d_wmma<5,5,5,5, 2,2,2,2, 8,8, 8,8, 0>
        <<<16 * 8 * 8, 256, 0, stream>>>(bufB, w3b, b3, bufA, nullptr);
    // Layer 4: (3,3,3,3) pad 1
    conv4d_wmma<3,3,3,3, 1,1,1,1, 8,8, 8,8, 0>
        <<<16 * 8 * 8, 256, 0, stream>>>(bufA, w4b, b4, bufB, nullptr);
    // Layer 5: (8,8,1,1) pad 0 -> (1,1,16,16), sigmoid, f32 out
    conv4d_wmma<8,8,1,1, 0,0,0,0, 8,8, 1,1, 1>
        <<<16 * 1 * 1, 256, 0, stream>>>(bufB, w5b, b5, nullptr, (float*)d_out);
}
